// SparseKMoE_1932735284124
// MI455X (gfx1250) — compile-verified, matrix operand order checked by reading, not run on hardware
//
#include <hip/hip_runtime.h>
#include <hip/hip_bf16.h>

// ---------------- problem constants (from reference) ----------------
#define M_TOK 4096          // B*T = 4*1024 tokens
#define DDIM  1024          // model dim
#define HDIM  2048          // hidden dim
#define NEXP  16            // experts
#define TOPK  3             // int(16*0.2)
#define NPAIR (M_TOK * TOPK)

#define KSTEP 32            // bf16 WMMA K
#define BM    128           // block tile M (tokens): 2 wave-rows x 64
#define BN    256           // block tile N: 4 wave-cols x 64
#define MT    4             // M sub-tiles per wave (4 x 16 = 64 rows)
#define NT    4             // N sub-tiles per wave (4 x 16 = 64 cols)

typedef __attribute__((ext_vector_type(16))) __bf16 v16bf;
typedef __attribute__((ext_vector_type(8)))  float  v8f;

union Frag { v16bf v; uint4 q[2]; };

__device__ __forceinline__ unsigned short f2bf(float f) {
    union { float f; unsigned u; } v; v.f = f;
    unsigned r = v.u + 0x7fffu + ((v.u >> 16) & 1u);   // round-to-nearest-even
    return (unsigned short)(r >> 16);
}
__device__ __forceinline__ unsigned pack2bf(float a, float b) {
    return (unsigned)f2bf(a) | ((unsigned)f2bf(b) << 16);
}

// ---------------- 0a) x: fp32 -> bf16, row-major (A is M x K, K contiguous) -------
__global__ __launch_bounds__(256)
void convert_x(const float* __restrict__ in, unsigned short* __restrict__ out) {
    size_t i = (size_t)blockIdx.x * 256 + threadIdx.x;      // one float4 per thread
    float4 v = *(const float4*)&in[i * 4];
    uint2 pk;
    pk.x = pack2bf(v.x, v.y);
    pk.y = pack2bf(v.z, v.w);
    *(uint2*)&out[i * 4] = pk;
}

// ---------------- 0b) weights: fp32 [R][C] -> bf16 [C][R] per expert --------------
// Produces N-major / K-contiguous panels so WMMA B fragments are two aligned
// 16B runs loadable directly with global_load_b128.
__global__ __launch_bounds__(256)
void transpose_convert(const float* __restrict__ in, unsigned short* __restrict__ out,
                       int R, int C) {
    const float* inp = in + (size_t)blockIdx.z * R * C;
    unsigned short* outp = out + (size_t)blockIdx.z * R * C;
    const int r0 = blockIdx.y * 32, c0 = blockIdx.x * 32;
    __shared__ unsigned short tile[32][33];                 // +1 pad vs bank conflicts
    const int t = threadIdx.x;
#pragma unroll
    for (int i = 0; i < 4; ++i) {                           // coalesced fp32 reads
        int idx = t + i * 256;
        int r = idx >> 5, c = idx & 31;
        tile[r][c] = f2bf(inp[(size_t)(r0 + r) * C + c0 + c]);
    }
    __syncthreads();
    const int c  = t >> 3;                                  // 32 columns
    const int r4 = (t & 7) * 4;                             // 4 consecutive rows
    uint2 pk;
    pk.x = (unsigned)tile[r4 + 0][c] | ((unsigned)tile[r4 + 1][c] << 16);
    pk.y = (unsigned)tile[r4 + 2][c] | ((unsigned)tile[r4 + 3][c] << 16);
    *(uint2*)&outp[(size_t)(c0 + c) * R + r0 + r4] = pk;    // coalesced bf16 writes
}

// ---------------- 1) gating: logits -> softmax -> top3 -> weights ----------------
__global__ __launch_bounds__(256)
void moe_gate(const float* __restrict__ x, const float* __restrict__ gate,
              int* __restrict__ counts, int* __restrict__ topk_e,
              float* __restrict__ topk_w) {
    const int wave = threadIdx.x >> 5;
    const int lane = threadIdx.x & 31;
    const int t = blockIdx.x * 8 + wave;
    if (t >= M_TOK) return;

    float acc[NEXP];
#pragma unroll
    for (int e = 0; e < NEXP; ++e) acc[e] = 0.f;

    const float* xr = x + (size_t)t * DDIM;
    for (int d = lane; d < DDIM; d += 32) {
        float xv = xr[d];
        const float4* g4 = (const float4*)(gate + (size_t)d * NEXP);
#pragma unroll
        for (int q = 0; q < 4; ++q) {
            float4 gv = g4[q];
            acc[q*4+0] += xv * gv.x; acc[q*4+1] += xv * gv.y;
            acc[q*4+2] += xv * gv.z; acc[q*4+3] += xv * gv.w;
        }
    }
#pragma unroll
    for (int e = 0; e < NEXP; ++e) {
#pragma unroll
        for (int s = 16; s > 0; s >>= 1) acc[e] += __shfl_xor(acc[e], s, 32);
    }
    if (lane == 0) {
        float mx = acc[0];
#pragma unroll
        for (int e = 1; e < NEXP; ++e) mx = fmaxf(mx, acc[e]);
        float p[NEXP]; float sum = 0.f;
#pragma unroll
        for (int e = 0; e < NEXP; ++e) { p[e] = __expf(acc[e] - mx); sum += p[e]; }
        float inv = 1.f / sum;
#pragma unroll
        for (int e = 0; e < NEXP; ++e) p[e] *= inv;

        int ids[TOPK]; float pv[TOPK]; float wsum = 0.f;
#pragma unroll
        for (int j = 0; j < TOPK; ++j) {
            int best = 0; float bv = p[0];
#pragma unroll
            for (int e = 1; e < NEXP; ++e) if (p[e] > bv) { bv = p[e]; best = e; }
            ids[j] = best; pv[j] = bv; p[best] = -1.f; wsum += bv;
        }
        float invw = 1.f / fmaxf(wsum, 1e-12f);
#pragma unroll
        for (int j = 0; j < TOPK; ++j) {
            topk_e[t * TOPK + j] = ids[j];
            topk_w[t * TOPK + j] = pv[j] * invw;
            atomicAdd(&counts[ids[j]], 1);
        }
    }
}

// ---------------- 2) tiny exclusive scan over 16 expert counts ----------------
__global__ void moe_scan(const int* __restrict__ counts, int* __restrict__ offsets) {
    if (threadIdx.x == 0) {
        int s = 0;
        for (int e = 0; e < NEXP; ++e) { offsets[e] = s; s += counts[e]; }
    }
}

// ---------------- 3) compact (token, weight) pairs grouped by expert --------------
__global__ __launch_bounds__(256)
void moe_scatter(const int* __restrict__ topk_e, const float* __restrict__ topk_w,
                 const int* __restrict__ offsets, int* __restrict__ cursor,
                 int* __restrict__ tok_list, float* __restrict__ w_list) {
    int t = blockIdx.x * blockDim.x + threadIdx.x;
    if (t >= M_TOK) return;
#pragma unroll
    for (int j = 0; j < TOPK; ++j) {
        int e = topk_e[t * TOPK + j];
        int pos = atomicAdd(&cursor[e], 1);
        int p = offsets[e] + pos;
        tok_list[p] = t;
        w_list[p]   = topk_w[t * TOPK + j];
    }
}

// ---------------- 4) layer 1: hidden = relu(Xg @ W1_e + b1_e) ----------------
// 8 waves/block as 2 (M) x 4 (N); each wave computes a 64x64 register tile:
// 4 M-subtiles x 4 N-subtiles of 16x16 C tiles -> 16 WMMAs per 16 b128 loads
// per K-step. Fragments load straight from L2-resident bf16 panels; no LDS.
__global__ __launch_bounds__(256)
void moe_ffn1(const unsigned short* __restrict__ xbf,
              const unsigned short* __restrict__ w1t,
              const float* __restrict__ b1,
              const int* __restrict__ counts, const int* __restrict__ offsets,
              const int* __restrict__ tok_list, unsigned short* __restrict__ hidden) {
    const int e = blockIdx.z;
    const int cnt = counts[e];
    const int strip = blockIdx.y;
    if (strip * BM >= cnt) return;
    const int off = offsets[e];
    const int n0 = blockIdx.x * BN;

    const int lane = threadIdx.x & 31, w = threadIdx.x >> 5;
    const int mh = w >> 2, nq = w & 3;
    const int l16 = lane & 15;
    const int klo = (lane < 16) ? 0 : 8;
    const int rowbase = strip * BM + mh * 64;

    const unsigned short* aptr[MT];
#pragma unroll
    for (int mt = 0; mt < MT; ++mt) {
        int row = rowbase + mt * 16 + l16;
        int tok = tok_list[off + ((row < cnt) ? row : 0)];
        aptr[mt] = xbf + (size_t)tok * DDIM + klo;
    }
    const unsigned short* bptr[NT];
#pragma unroll
    for (int nt = 0; nt < NT; ++nt) {
        int n = n0 + nq * 64 + nt * 16 + l16;
        bptr[nt] = w1t + ((size_t)e * HDIM + n) * DDIM + klo;
    }

    v8f acc[MT][NT];
#pragma unroll
    for (int mt = 0; mt < MT; ++mt)
#pragma unroll
        for (int nt = 0; nt < NT; ++nt)
#pragma unroll
            for (int j = 0; j < 8; ++j) acc[mt][nt][j] = 0.f;

#pragma unroll 2
    for (int k0 = 0; k0 < DDIM; k0 += KSTEP) {
        Frag fa[MT], fb[NT];
#pragma unroll
        for (int mt = 0; mt < MT; ++mt) {
            fa[mt].q[0] = *(const uint4*)(aptr[mt] + k0);
            fa[mt].q[1] = *(const uint4*)(aptr[mt] + k0 + 16);
        }
#pragma unroll
        for (int nt = 0; nt < NT; ++nt) {
            fb[nt].q[0] = *(const uint4*)(bptr[nt] + k0);
            fb[nt].q[1] = *(const uint4*)(bptr[nt] + k0 + 16);
        }
#pragma unroll
        for (int mt = 0; mt < MT; ++mt)
#pragma unroll
            for (int nt = 0; nt < NT; ++nt)
                acc[mt][nt] = __builtin_amdgcn_wmma_f32_16x16x32_bf16(
                    false, fa[mt].v, false, fb[nt].v, (short)0, acc[mt][nt],
                    false, false);
    }

    const int mbase = (lane < 16) ? 0 : 8;
#pragma unroll
    for (int mt = 0; mt < MT; ++mt)
#pragma unroll
        for (int nt = 0; nt < NT; ++nt) {
            int n = n0 + nq * 64 + nt * 16 + l16;
            float bias = b1[e * HDIM + n];
#pragma unroll
            for (int g = 0; g < 8; ++g) {
                int row = rowbase + mt * 16 + mbase + g;
                if (row < cnt) {
                    float v = fmaxf(acc[mt][nt][g] + bias, 0.f);
                    hidden[(size_t)(off + row) * HDIM + n] = f2bf(v);
                }
            }
        }
}

// ---------------- 5) layer 2: out += w * (hidden @ W2_e + b2_e) ----------------
__global__ __launch_bounds__(256)
void moe_ffn2(const unsigned short* __restrict__ hidden,
              const unsigned short* __restrict__ w2t,
              const float* __restrict__ b2,
              const int* __restrict__ counts, const int* __restrict__ offsets,
              const int* __restrict__ tok_list, const float* __restrict__ w_list,
              float* __restrict__ out) {
    const int e = blockIdx.z;
    const int cnt = counts[e];
    const int strip = blockIdx.y;
    if (strip * BM >= cnt) return;
    const int off = offsets[e];
    const int n0 = blockIdx.x * BN;

    const int lane = threadIdx.x & 31, w = threadIdx.x >> 5;
    const int mh = w >> 2, nq = w & 3;
    const int l16 = lane & 15;
    const int klo = (lane < 16) ? 0 : 8;
    const int rowbase = strip * BM + mh * 64;

    const unsigned short* aptr[MT];
#pragma unroll
    for (int mt = 0; mt < MT; ++mt) {
        int row = rowbase + mt * 16 + l16;
        int rc = (row < cnt) ? row : 0;
        aptr[mt] = hidden + (size_t)(off + rc) * HDIM + klo;
    }
    const unsigned short* bptr[NT];
#pragma unroll
    for (int nt = 0; nt < NT; ++nt) {
        int n = n0 + nq * 64 + nt * 16 + l16;
        bptr[nt] = w2t + ((size_t)e * DDIM + n) * HDIM + klo;
    }

    v8f acc[MT][NT];
#pragma unroll
    for (int mt = 0; mt < MT; ++mt)
#pragma unroll
        for (int nt = 0; nt < NT; ++nt)
#pragma unroll
            for (int j = 0; j < 8; ++j) acc[mt][nt][j] = 0.f;

#pragma unroll 2
    for (int k0 = 0; k0 < HDIM; k0 += KSTEP) {
        Frag fa[MT], fb[NT];
#pragma unroll
        for (int mt = 0; mt < MT; ++mt) {
            fa[mt].q[0] = *(const uint4*)(aptr[mt] + k0);
            fa[mt].q[1] = *(const uint4*)(aptr[mt] + k0 + 16);
        }
#pragma unroll
        for (int nt = 0; nt < NT; ++nt) {
            fb[nt].q[0] = *(const uint4*)(bptr[nt] + k0);
            fb[nt].q[1] = *(const uint4*)(bptr[nt] + k0 + 16);
        }
#pragma unroll
        for (int mt = 0; mt < MT; ++mt)
#pragma unroll
            for (int nt = 0; nt < NT; ++nt)
                acc[mt][nt] = __builtin_amdgcn_wmma_f32_16x16x32_bf16(
                    false, fa[mt].v, false, fb[nt].v, (short)0, acc[mt][nt],
                    false, false);
    }

    const int mbase = (lane < 16) ? 0 : 8;
#pragma unroll
    for (int mt = 0; mt < MT; ++mt)
#pragma unroll
        for (int nt = 0; nt < NT; ++nt) {
            int n = n0 + nq * 64 + nt * 16 + l16;
            float bias = b2[e * DDIM + n];
#pragma unroll
            for (int g = 0; g < 8; ++g) {
                int row = rowbase + mt * 16 + mbase + g;
                if (row < cnt) {
                    int p = off + row;
                    int t = tok_list[p];
                    float wt = w_list[p];
                    atomicAdd(&out[(size_t)t * DDIM + n], wt * (acc[mt][nt][g] + bias));
                }
            }
        }
}

// ---------------- host side ----------------
extern "C" void kernel_launch(void* const* d_in, const int* in_sizes, int n_in,
                              void* d_out, int out_size, void* d_ws, size_t ws_size,
                              hipStream_t stream) {
    const float* x    = (const float*)d_in[0];
    const float* gate = (const float*)d_in[1];
    const float* W1   = (const float*)d_in[2];
    const float* b1   = (const float*)d_in[3];
    const float* W2   = (const float*)d_in[4];
    const float* b2   = (const float*)d_in[5];
    float* out = (float*)d_out;

    char* ws = (char*)d_ws;
    int*   counts   = (int*)(ws + 0);
    int*   cursor   = (int*)(ws + 64);
    int*   offsets  = (int*)(ws + 128);
    int*   topk_e   = (int*)(ws + 256);
    float* topk_w   = (float*)(ws + 256 + 4  * (size_t)NPAIR);
    int*   tok_list = (int*)(ws + 256 + 8  * (size_t)NPAIR);
    float* w_list   = (float*)(ws + 256 + 12 * (size_t)NPAIR);
    size_t base = 256 + 16 * (size_t)NPAIR;                    // 16B aligned
    unsigned short* xbf    = (unsigned short*)(ws + base);
    base += (size_t)M_TOK * DDIM * 2;                          // 8 MB
    unsigned short* w1t    = (unsigned short*)(ws + base);
    base += (size_t)NEXP * DDIM * HDIM * 2;                    // 64 MB
    unsigned short* w2t    = (unsigned short*)(ws + base);
    base += (size_t)NEXP * HDIM * DDIM * 2;                    // 64 MB
    unsigned short* hidden = (unsigned short*)(ws + base);     // 48 MB

    hipMemsetAsync(counts, 0, 128, stream);                    // counts + cursor
    hipMemsetAsync(out, 0, (size_t)out_size * sizeof(float), stream);

    // precision prep: one-shot bf16 conversion + weight transposition
    convert_x<<<(M_TOK * DDIM) / (256 * 4), 256, 0, stream>>>(x, xbf);
    dim3 t1(HDIM / 32, DDIM / 32, NEXP);                       // W1: [D][H] -> [H][D]
    transpose_convert<<<t1, 256, 0, stream>>>(W1, w1t, DDIM, HDIM);
    dim3 t2(DDIM / 32, HDIM / 32, NEXP);                       // W2: [H][D] -> [D][H]
    transpose_convert<<<t2, 256, 0, stream>>>(W2, w2t, HDIM, DDIM);

    // routing
    moe_gate   <<<M_TOK / 8,   256, 0, stream>>>(x, gate, counts, topk_e, topk_w);
    moe_scan   <<<1,            32, 0, stream>>>(counts, offsets);
    moe_scatter<<<M_TOK / 256, 256, 0, stream>>>(topk_e, topk_w, offsets, cursor,
                                                 tok_list, w_list);

    // expert-grouped WMMA GEMMs
    dim3 g1(HDIM / BN, M_TOK / BM, NEXP);
    moe_ffn1<<<g1, 256, 0, stream>>>(xbf, w1t, b1, counts, offsets, tok_list, hidden);
    dim3 g2(DDIM / BN, M_TOK / BM, NEXP);
    moe_ffn2<<<g2, 256, 0, stream>>>(hidden, w2t, b2, counts, offsets, tok_list,
                                     w_list, out);
}
